// Attention2_7138235646706
// MI455X (gfx1250) — compile-verified, matrix operand order checked by reading, not run on hardware
//
#include <hip/hip_runtime.h>

// ---------------------------------------------------------------------------
// Query-axis-softmax MHA for MI455X (gfx1250), bf16 WMMA, flash-style 2-pass.
// B=2, N=4096, D=256, H=8, dh=32.  Never materializes the 1 GB score matrix.
// Round 2: 32x32 tiles per wave (2x data reuse), scale*log2e folded into Q.
// ---------------------------------------------------------------------------

#define BSZ    2
#define NSEQ   4096
#define DMODEL 256
#define NHEAD  8
#define DHEAD  32
#define MTOT   (BSZ * NSEQ)          // 8192 rows
#define SCALE  0.17677669529663687f  // 1/sqrt(32)
#define LOG2E  1.4426950408889634f
#define QSCALE (SCALE * LOG2E)       // folded into Q projection -> log2 domain

typedef __bf16 bf16_t;
typedef __attribute__((ext_vector_type(16))) __bf16 bf16x16;
typedef __attribute__((ext_vector_type(8)))  __bf16 bf16x8;
typedef __attribute__((ext_vector_type(8)))  float  f32x8;

// ---- WMMA wrapper ----------------------------------------------------------
__device__ __forceinline__ f32x8 wmma_bf16(bf16x16 a, bf16x16 b, f32x8 c) {
  return __builtin_amdgcn_wmma_f32_16x16x32_bf16(
      /*neg_a=*/false, a, /*neg_b=*/false, b,
      /*c_mod=*/(short)0, c, /*reuse_a=*/false, /*reuse_b=*/false);
}

// ---- Fragment load (A-layout; also serves B when storage is K-contiguous) --
// lane r = lane&15 (row for A / column for B), g = lane>>4.
// Element j<8  -> K = 8g + j ; element j>=8 -> K = 16 + 8g + (j-8).
// Two aligned 16-byte loads per lane.
__device__ __forceinline__ bf16x16 ldfrag(const bf16_t* base, int ld) {
  int lane = threadIdx.x & 31;
  int r = lane & 15;
  int g = (lane >> 4) & 1;
  const bf16_t* p = base + (size_t)r * ld + 8 * g;
  bf16x8 lo = *(const bf16x8*)(p);
  bf16x8 hi = *(const bf16x8*)(p + 16);
  bf16x16 f;
#pragma unroll
  for (int i = 0; i < 8; ++i) { f[i] = lo[i]; f[i + 8] = hi[i]; }
  return f;
}

// ---- per-key-column online softmax update (log2 domain) --------------------
__device__ __forceinline__ void online_update(const f32x8& acc, float& mrun, float& zrun) {
  float sc[8];
  float tmax = -1e30f;
#pragma unroll
  for (int r = 0; r < 8; ++r) { sc[r] = acc[r]; tmax = fmaxf(tmax, sc[r]); }
  tmax = fmaxf(tmax, __shfl_xor(tmax, 16));     // combine lane halves (rows)
  float mnew = fmaxf(mrun, tmax);
  float ssum = 0.f;
#pragma unroll
  for (int r = 0; r < 8; ++r) ssum += exp2f(sc[r] - mnew);
  ssum += __shfl_xor(ssum, 16);
  zrun = zrun * exp2f(mrun - mnew) + ssum;
  mrun = mnew;
}

// ---- f32 -> bf16 conversion ------------------------------------------------
__global__ void k_cvt_bf16(const float* __restrict__ s, bf16_t* __restrict__ d, int n) {
  int i = blockIdx.x * blockDim.x + threadIdx.x;
  int stride = gridDim.x * blockDim.x;
  for (; i < n; i += stride) d[i] = (bf16_t)s[i];
}

// ---- QKV projections: out[b][h][n][dh] bf16, C = (x @ W^T + bias)*oscale ---
// grid(x = MTOT/128, y = DMODEL/32, z = 3), block 128: wave = 32x32 tile.
__global__ void k_proj_qkv(const bf16_t* __restrict__ xb,
                           const bf16_t* __restrict__ wq, const bf16_t* __restrict__ wk,
                           const bf16_t* __restrict__ wv,
                           const float* __restrict__ bq, const float* __restrict__ bk,
                           const float* __restrict__ bv,
                           bf16_t* __restrict__ Qb, bf16_t* __restrict__ Kb,
                           bf16_t* __restrict__ Vb) {
  int wave = threadIdx.x >> 5;
  int m0 = (blockIdx.x * 4 + wave) * 32;          // 32 output rows
  int j0 = blockIdx.y * 32;                        // 32 output cols (one head)
  const bf16_t* W; const float* bias; bf16_t* out; float oscale;
  if (blockIdx.z == 0)      { W = wq; bias = bq; out = Qb; oscale = QSCALE; }
  else if (blockIdx.z == 1) { W = wk; bias = bk; out = Kb; oscale = 1.f; }
  else                      { W = wv; bias = bv; out = Vb; oscale = 1.f; }

  f32x8 acc00 = {}, acc01 = {}, acc10 = {}, acc11 = {};
#pragma unroll
  for (int k0 = 0; k0 < DMODEL; k0 += 32) {
    bf16x16 a0 = ldfrag(xb + (size_t)m0 * DMODEL + k0, DMODEL);
    bf16x16 a1 = ldfrag(xb + (size_t)(m0 + 16) * DMODEL + k0, DMODEL);
    bf16x16 b0 = ldfrag(W + (size_t)j0 * DMODEL + k0, DMODEL);
    bf16x16 b1 = ldfrag(W + (size_t)(j0 + 16) * DMODEL + k0, DMODEL);
    acc00 = wmma_bf16(a0, b0, acc00);
    acc01 = wmma_bf16(a0, b1, acc01);
    acc10 = wmma_bf16(a1, b0, acc10);
    acc11 = wmma_bf16(a1, b1, acc11);
  }
  int lane = threadIdx.x & 31;
  int nn = lane & 15, g = lane >> 4;
  int h = j0 >> 5;
  int c0 = j0 + nn, c1 = j0 + 16 + nn;
#pragma unroll
  for (int r = 0; r < 8; ++r) {
    int mA = m0 + r + 8 * g;
    int mB = mA + 16;
    int bA = mA >> 12, qA = mA & (NSEQ - 1);
    int bB = mB >> 12, qB = mB & (NSEQ - 1);
    out[((size_t)(bA * NHEAD + h) * NSEQ + qA) * DHEAD + (c0 & 31)] =
        (bf16_t)((acc00[r] + bias[c0]) * oscale);
    out[((size_t)(bA * NHEAD + h) * NSEQ + qA) * DHEAD + (c1 & 31)] =
        (bf16_t)((acc01[r] + bias[c1]) * oscale);
    out[((size_t)(bB * NHEAD + h) * NSEQ + qB) * DHEAD + (c0 & 31)] =
        (bf16_t)((acc10[r] + bias[c0]) * oscale);
    out[((size_t)(bB * NHEAD + h) * NSEQ + qB) * DHEAD + (c1 & 31)] =
        (bf16_t)((acc11[r] + bias[c1]) * oscale);
  }
}

// ---- Pass 1: per-key-column online softmax stats m[k], Z[k] ----------------
// One wave owns 32 key columns of one (b,h); streams all 256 query tiles.
// Each Q A-fragment load is amortized over 2 WMMAs.
__global__ void k_attn_stats(const bf16_t* __restrict__ Qb, const bf16_t* __restrict__ Kb,
                             float* __restrict__ mbuf, float* __restrict__ zbuf) {
  int wave = threadIdx.x >> 5;
  int id = blockIdx.x * 4 + wave;                 // 2048 waves total
  int bh = id >> 7;                               // 16 (b,h) pairs
  int kb = id & 127;                              // 32-key block
  int k0 = kb * 32;
  const bf16_t* Qh = Qb + (size_t)bh * NSEQ * DHEAD;
  const bf16_t* Kh = Kb + (size_t)bh * NSEQ * DHEAD;

  bf16x16 bf0 = ldfrag(Kh + (size_t)k0 * DHEAD, DHEAD);         // keys k0..k0+15
  bf16x16 bf1 = ldfrag(Kh + (size_t)(k0 + 16) * DHEAD, DHEAD);  // keys +16..+31

  float m0r = -1e30f, z0r = 0.f, m1r = -1e30f, z1r = 0.f;
  for (int q0 = 0; q0 < NSEQ; q0 += 16) {
    bf16x16 a = ldfrag(Qh + (size_t)q0 * DHEAD, DHEAD);
    f32x8 acc0 = {}, acc1 = {};
    acc0 = wmma_bf16(a, bf0, acc0);
    acc1 = wmma_bf16(a, bf1, acc1);
    online_update(acc0, m0r, z0r);
    online_update(acc1, m1r, z1r);
  }
  int lane = threadIdx.x & 31;
  if (lane < 16) {
    mbuf[(size_t)bh * NSEQ + k0 + lane]      = m0r;
    zbuf[(size_t)bh * NSEQ + k0 + lane]      = z0r;
    mbuf[(size_t)bh * NSEQ + k0 + 16 + lane] = m1r;
    zbuf[(size_t)bh * NSEQ + k0 + 16 + lane] = z1r;
  }
}

// ---- V' = V / Z, transposed to d-major [bh][dh][N] for B-fragment loads ----
__global__ void k_build_vt(const bf16_t* __restrict__ Vb, const float* __restrict__ zbuf,
                           bf16_t* __restrict__ Vtb) {
  int i = blockIdx.x * blockDim.x + threadIdx.x;
  int stride = gridDim.x * blockDim.x;
  int total = BSZ * NHEAD * NSEQ * DHEAD;
  for (; i < total; i += stride) {
    int bh = i / (NSEQ * DHEAD);
    int rem = i - bh * (NSEQ * DHEAD);
    int k = rem >> 5, dd = rem & 31;
    float v = (float)Vb[i] / zbuf[(size_t)bh * NSEQ + k];
    Vtb[((size_t)bh * DHEAD + dd) * NSEQ + k] = (bf16_t)v;
  }
}

// ---- Pass 2: ctx[q,:] = sum_k 2^(s-m[k]) * V'[k,:] -------------------------
// One wave owns 32 queries (two 16-row tiles) of one (b,h); streams keys in
// blocks of 32.  K-tile and V'-tile fragments are each amortized over 2 WMMAs.
// P tiles re-laid C->A through wave-private LDS (uniform trip counts make
// block-level __syncthreads safe).
__global__ void k_attn_ctx(const bf16_t* __restrict__ Qb, const bf16_t* __restrict__ Kb,
                           const bf16_t* __restrict__ Vtb, const float* __restrict__ mbuf,
                           bf16_t* __restrict__ ctxb) {
  __shared__ bf16_t Pbuf[4][2][16 * 32];          // 8 KB
  int wave = threadIdx.x >> 5;
  int lane = threadIdx.x & 31;
  int id = blockIdx.x * 4 + wave;                 // 2048 waves total
  int bh = id >> 7;
  int qt = id & 127;
  int q0 = qt * 32;
  const bf16_t* Qh = Qb + (size_t)bh * NSEQ * DHEAD;
  const bf16_t* Kh = Kb + (size_t)bh * NSEQ * DHEAD;
  const bf16_t* Vt = Vtb + (size_t)bh * DHEAD * NSEQ;
  const float*  mh = mbuf + (size_t)bh * NSEQ;
  bf16_t* P0 = Pbuf[wave][0];
  bf16_t* P1 = Pbuf[wave][1];

  bf16x16 af0 = ldfrag(Qh + (size_t)q0 * DHEAD, DHEAD);         // q0..q0+15
  bf16x16 af1 = ldfrag(Qh + (size_t)(q0 + 16) * DHEAD, DHEAD);  // +16..+31
  f32x8 acc00 = {}, acc01 = {}, acc10 = {}, acc11 = {};
  int nn = lane & 15, g = lane >> 4;

  for (int k0 = 0; k0 < NSEQ; k0 += 32) {
#pragma unroll
    for (int half = 0; half < 2; ++half) {
      int kk = k0 + half * 16;
      bf16x16 bfrag = ldfrag(Kh + (size_t)kk * DHEAD, DHEAD);
      f32x8 s0 = {}, s1 = {};
      s0 = wmma_bf16(af0, bfrag, s0);
      s1 = wmma_bf16(af1, bfrag, s1);
      float mcol = mh[kk + nn];                   // per-key-column max (log2)
#pragma unroll
      for (int r = 0; r < 8; ++r) {
        P0[(r + 8 * g) * 32 + half * 16 + nn] = (bf16_t)exp2f(s0[r] - mcol);
        P1[(r + 8 * g) * 32 + half * 16 + nn] = (bf16_t)exp2f(s1[r] - mcol);
      }
    }
    __syncthreads();                              // P visible as A-fragment src
    bf16x16 pa0 = ldfrag(P0, 32);                 // A: 16(q) x 32(keys)
    bf16x16 pa1 = ldfrag(P1, 32);
    bf16x16 vb0 = ldfrag(Vt + k0, NSEQ);          // B cols d=0..15
    bf16x16 vb1 = ldfrag(Vt + (size_t)16 * NSEQ + k0, NSEQ);  // d=16..31
    acc00 = wmma_bf16(pa0, vb0, acc00);
    acc01 = wmma_bf16(pa0, vb1, acc01);
    acc10 = wmma_bf16(pa1, vb0, acc10);
    acc11 = wmma_bf16(pa1, vb1, acc11);
    __syncthreads();
  }

  int b = bh >> 3, h = bh & 7;
#pragma unroll
  for (int r = 0; r < 8; ++r) {
    int qA = q0 + r + 8 * g;
    int qB = qA + 16;
    size_t rowA = (size_t)(b * NSEQ + qA) * DMODEL + h * DHEAD;
    size_t rowB = (size_t)(b * NSEQ + qB) * DMODEL + h * DHEAD;
    ctxb[rowA + nn]      = (bf16_t)acc00[r];
    ctxb[rowA + 16 + nn] = (bf16_t)acc01[r];
    ctxb[rowB + nn]      = (bf16_t)acc10[r];
    ctxb[rowB + 16 + nn] = (bf16_t)acc11[r];
  }
}

// ---- Output projection: out = ctx @ Wo^T + bo (f32 out), 32x32 per wave ----
__global__ void k_proj_out(const bf16_t* __restrict__ ctxb, const bf16_t* __restrict__ wo,
                           const float* __restrict__ bo, float* __restrict__ out) {
  int wave = threadIdx.x >> 5;
  int m0 = (blockIdx.x * 4 + wave) * 32;
  int j0 = blockIdx.y * 32;
  f32x8 acc00 = {}, acc01 = {}, acc10 = {}, acc11 = {};
#pragma unroll
  for (int k0 = 0; k0 < DMODEL; k0 += 32) {
    bf16x16 a0 = ldfrag(ctxb + (size_t)m0 * DMODEL + k0, DMODEL);
    bf16x16 a1 = ldfrag(ctxb + (size_t)(m0 + 16) * DMODEL + k0, DMODEL);
    bf16x16 b0 = ldfrag(wo + (size_t)j0 * DMODEL + k0, DMODEL);
    bf16x16 b1 = ldfrag(wo + (size_t)(j0 + 16) * DMODEL + k0, DMODEL);
    acc00 = wmma_bf16(a0, b0, acc00);
    acc01 = wmma_bf16(a0, b1, acc01);
    acc10 = wmma_bf16(a1, b0, acc10);
    acc11 = wmma_bf16(a1, b1, acc11);
  }
  int lane = threadIdx.x & 31;
  int nn = lane & 15, g = lane >> 4;
#pragma unroll
  for (int r = 0; r < 8; ++r) {
    int mA = m0 + r + 8 * g;
    int mB = mA + 16;
    out[(size_t)mA * DMODEL + j0 + nn]      = acc00[r] + bo[j0 + nn];
    out[(size_t)mA * DMODEL + j0 + 16 + nn] = acc01[r] + bo[j0 + 16 + nn];
    out[(size_t)mB * DMODEL + j0 + nn]      = acc10[r] + bo[j0 + nn];
    out[(size_t)mB * DMODEL + j0 + 16 + nn] = acc11[r] + bo[j0 + 16 + nn];
  }
}

// ---------------------------------------------------------------------------
extern "C" void kernel_launch(void* const* d_in, const int* in_sizes, int n_in,
                              void* d_out, int out_size, void* d_ws, size_t ws_size,
                              hipStream_t stream) {
  const float* x  = (const float*)d_in[0];
  const float* Wq = (const float*)d_in[1];
  const float* bq = (const float*)d_in[2];
  const float* Wk = (const float*)d_in[3];
  const float* bk = (const float*)d_in[4];
  const float* Wv = (const float*)d_in[5];
  const float* bv = (const float*)d_in[6];
  const float* Wo = (const float*)d_in[7];
  const float* bo = (const float*)d_in[8];
  float* out = (float*)d_out;

  char* ws = (char*)d_ws;
  size_t off = 0;
  auto take = [&](size_t bytes) -> void* {
    void* p = ws + off;
    off += (bytes + 255) & ~(size_t)255;
    return p;
  };
  bf16_t* xb   = (bf16_t*)take((size_t)MTOT * DMODEL * 2);
  bf16_t* wqb  = (bf16_t*)take((size_t)DMODEL * DMODEL * 2);
  bf16_t* wkb  = (bf16_t*)take((size_t)DMODEL * DMODEL * 2);
  bf16_t* wvb  = (bf16_t*)take((size_t)DMODEL * DMODEL * 2);
  bf16_t* wob  = (bf16_t*)take((size_t)DMODEL * DMODEL * 2);
  bf16_t* Qb   = (bf16_t*)take((size_t)MTOT * DMODEL * 2);
  bf16_t* Kb   = (bf16_t*)take((size_t)MTOT * DMODEL * 2);
  bf16_t* Vb   = (bf16_t*)take((size_t)MTOT * DMODEL * 2);
  float*  mbuf = (float*)take((size_t)BSZ * NHEAD * NSEQ * 4);
  float*  zbuf = (float*)take((size_t)BSZ * NHEAD * NSEQ * 4);
  bf16_t* Vtb  = (bf16_t*)take((size_t)MTOT * DMODEL * 2);
  bf16_t* ctxb = (bf16_t*)take((size_t)MTOT * DMODEL * 2);

  // 1) f32 -> bf16 conversions (W kept row-major: already K-contiguous for B).
  k_cvt_bf16<<<1024, 256, 0, stream>>>(x,  xb,  MTOT * DMODEL);
  k_cvt_bf16<<<64,   256, 0, stream>>>(Wq, wqb, DMODEL * DMODEL);
  k_cvt_bf16<<<64,   256, 0, stream>>>(Wk, wkb, DMODEL * DMODEL);
  k_cvt_bf16<<<64,   256, 0, stream>>>(Wv, wvb, DMODEL * DMODEL);
  k_cvt_bf16<<<64,   256, 0, stream>>>(Wo, wob, DMODEL * DMODEL);

  // 2) Q/K/V projections into per-head layout (Q pre-scaled to log2 domain).
  k_proj_qkv<<<dim3(MTOT / 128, DMODEL / 32, 3), 128, 0, stream>>>(
      xb, wqb, wkb, wvb, bq, bk, bv, Qb, Kb, Vb);

  // 3) Per-key-column softmax stats (streaming, scores never stored).
  k_attn_stats<<<512, 128, 0, stream>>>(Qb, Kb, mbuf, zbuf);

  // 4) V' = V/Z, d-major transpose.
  k_build_vt<<<1024, 256, 0, stream>>>(Vb, zbuf, Vtb);

  // 5) Context pass (recompute S tiles, P@V').
  k_attn_ctx<<<512, 128, 0, stream>>>(Qb, Kb, Vtb, mbuf, ctxb);

  // 6) Output projection (f32).
  k_proj_out<<<dim3(MTOT / 128, DMODEL / 32), 128, 0, stream>>>(ctxb, wob, bo, out);
}
